// EvolutionForSequence_54494545051809
// MI455X (gfx1250) — matrix operand order repaired and checked by reading, not compiled, not run on hardware
//
#include <hip/hip_runtime.h>
#include <hip/hip_bf16.h>

// ---------------- model constants ----------------
#define BB 4
#define LL 2048
#define DD 512
#define HH 8
#define HD 64
#define MM (BB * LL)   // 8192 rows

typedef __attribute__((ext_vector_type(16))) __bf16 v16bf;
typedef __attribute__((ext_vector_type(8)))  float  v8f;

static __device__ inline v8f zero_v8() {
    v8f z;
#pragma unroll
    for (int i = 0; i < 8; i++) z[i] = 0.f;
    return z;
}

// ---- fp32x2 -> packed bf16x2 (RNE) ----
static __device__ inline unsigned pack_bf16x2_manual(float x, float y) {
    unsigned ux = __builtin_bit_cast(unsigned, x);
    unsigned uy = __builtin_bit_cast(unsigned, y);
    ux = ux + 0x7fffu + ((ux >> 16) & 1u);
    uy = uy + 0x7fffu + ((uy >> 16) & 1u);
    return (ux >> 16) | (uy & 0xffff0000u);
}
template <typename T>
static __device__ inline unsigned as_u32_or_manual(T v, float x, float y) {
    if constexpr (sizeof(T) == sizeof(unsigned)) {
        return __builtin_bit_cast(unsigned, v);
    } else {
        return pack_bf16x2_manual(x, y);
    }
}
static __device__ inline unsigned pack_bf16x2(float x, float y) {
#if __has_builtin(__builtin_amdgcn_cvt_pk_bf16_f32)
    return as_u32_or_manual(__builtin_amdgcn_cvt_pk_bf16_f32(x, y), x, y);
#else
    return pack_bf16x2_manual(x, y);
#endif
}
static __device__ inline unsigned short f2bf_bits(float f) {
    unsigned u = __builtin_bit_cast(unsigned, f);
    return (unsigned short)((u + 0x7fffu + ((u >> 16) & 1u)) >> 16);
}

union FragU {
    uint4    q[2];
    unsigned u[8];
    v16bf    v;
};

// =====================================================================
// Generic WMMA GEMM:  C[M x N] = epilogue( sum_seg A_seg @ W + bias )
//  - A: up to 4 segments, each 512 cols wide, common row stride lda.
//  - K loop: outer over 512-wide segments (fixed pointers, no per-step
//    select chain), inner 16 K-steps of 32.
//  - LDS holds bf16: A as [row][k-pairs] stride 20 uints,
//    W transposed as [col][k-pairs] stride 20 uints.  All fragment
//    reads are 16B-aligned ds_load_b128, conversion done once at stage.
//  - tile 128x128, 8 waves, each wave 64x32 = 4x2 wmma tiles.
// =====================================================================
__global__ __launch_bounds__(256) void gemm_wmma_kernel(
    const float* __restrict__ A0, const float* __restrict__ A1,
    const float* __restrict__ A2, const float* __restrict__ A3,
    int lda, const float* __restrict__ W, int N, int K,
    const float* __restrict__ bias, const float* __restrict__ residual,
    float* __restrict__ C, int act)
{
    __shared__ unsigned AsU[128 * 20];  // [row][kpair 0..15], pad to 20
    __shared__ unsigned BsU[128 * 20];  // [col][kpair 0..15], pad to 20

    const int tid  = threadIdx.x;
    const int wid  = tid >> 5;
    const int lane = tid & 31;
    const int half = lane >> 4;
    const int l16  = lane & 15;

    const int tileN = blockIdx.x * 128;
    const int tileM = blockIdx.y * 128;
    const int waveM = (wid & 1) * 64;
    const int waveN = (wid >> 1) * 32;

    v8f acc[4][2];
#pragma unroll
    for (int i = 0; i < 4; i++)
#pragma unroll
        for (int j = 0; j < 2; j++) acc[i][j] = zero_v8();

    const int nseg = K >> 9;                 // K is a multiple of 512
    for (int seg = 0; seg < nseg; seg++) {
        const float* Ap = (seg == 0) ? A0 : (seg == 1) ? A1 : (seg == 2) ? A2 : A3;
        const float* Wseg = W + ((size_t)seg << 9) * N;
        const bool moreK = (seg + 1 < nseg);

        for (int kc = 0; kc < 512; kc += 32) {
            __syncthreads();
            // ---- stage A tile 128x32 -> bf16 pairs ----
            {
                const int r  = tid >> 3;          // 0..31
                const int c4 = (tid & 7) * 4;     // float col 0..28
#pragma unroll
                for (int rr = 0; rr < 128; rr += 32) {
                    const float* src = Ap + (size_t)(tileM + rr + r) * lda + kc + c4;
                    if (moreK || kc + 32 < 512) __builtin_prefetch(src + 32, 0, 1);
                    const float4 v = *(const float4*)src;
                    uint2 p;
                    p.x = pack_bf16x2(v.x, v.y);
                    p.y = pack_bf16x2(v.z, v.w);
                    *(uint2*)&AsU[(rr + r) * 20 + (c4 >> 1)] = p;  // ds_store_b64
                }
            }
            // ---- stage W tile 32x128 -> transposed bf16 pairs ----
            {
                const int r  = tid >> 5;          // 0..7
                const int c4 = lane * 4;          // col 0..124
#pragma unroll
                for (int pp = 0; pp < 2; pp++) {
                    const int kp = pp * 8 + r;    // k-pair 0..15
                    const float* src0 = Wseg + (size_t)(kc + 2 * kp) * N + tileN + c4;
                    if (moreK || kc + 32 < 512)
                        __builtin_prefetch(src0 + (size_t)32 * N, 0, 1);
                    const float4 v0 = *(const float4*)src0;
                    const float4 v1 = *(const float4*)(src0 + N);
                    BsU[(c4 + 0) * 20 + kp] = pack_bf16x2(v0.x, v1.x);
                    BsU[(c4 + 1) * 20 + kp] = pack_bf16x2(v0.y, v1.y);
                    BsU[(c4 + 2) * 20 + kp] = pack_bf16x2(v0.z, v1.z);
                    BsU[(c4 + 3) * 20 + kp] = pack_bf16x2(v0.w, v1.w);
                }
            }
            __syncthreads();

            // ---- B fragments: 2x ds_load_b128, no conversion ----
            v16bf bfrag[2];
#pragma unroll
            for (int j = 0; j < 2; j++) {
                const int col = waveN + j * 16 + l16;
                FragU f;
                f.q[0] = *(const uint4*)&BsU[col * 20 + half * 4 + 0];
                f.q[1] = *(const uint4*)&BsU[col * 20 + half * 4 + 8];
                bfrag[j] = f.v;
            }
            // ---- A fragments + WMMA ----
#pragma unroll
            for (int i = 0; i < 4; i++) {
                const int row = waveM + i * 16 + l16;
                FragU f;
                f.q[0] = *(const uint4*)&AsU[row * 20 + half * 4 + 0];
                f.q[1] = *(const uint4*)&AsU[row * 20 + half * 4 + 8];
                acc[i][0] = __builtin_amdgcn_wmma_f32_16x16x32_bf16(
                    false, f.v, false, bfrag[0], (short)0, acc[i][0], false, false);
                acc[i][1] = __builtin_amdgcn_wmma_f32_16x16x32_bf16(
                    false, f.v, false, bfrag[1], (short)0, acc[i][1], false, false);
            }
        }
    }

    // ---- epilogue (C layout: lane=n, vgpr r -> m = r + 8*half) ----
#pragma unroll
    for (int i = 0; i < 4; i++) {
        const int gr = tileM + waveM + i * 16 + 8 * half;
#pragma unroll
        for (int j = 0; j < 2; j++) {
            const int gc = tileN + waveN + j * 16 + l16;
            const float bv = bias ? bias[gc] : 0.f;
#pragma unroll
            for (int r = 0; r < 8; r++) {
                float v = acc[i][j][r] + bv;
                if (act == 1)      v = v / (1.f + __expf(-v));   // silu
                else if (act == 2) v = tanhf(v);
                const size_t idx = (size_t)(gr + r) * N + gc;
                if (residual) v += residual[idx];
                C[idx] = v;
            }
        }
    }
}

// =====================================================================
// Fused causal sigmoid-attention.  Q,K,V,Ctx are [B,L,H,HD] fp32.
// One wave per 16-row Q tile; scores & ctx via bf16 WMMA; P staged
// through LDS (bf16, stride-20 padded) -> b128 re-fragment loads.
// =====================================================================
__global__ __launch_bounds__(256) void attn_kernel(
    const float* __restrict__ Qb, const float* __restrict__ Kb,
    const float* __restrict__ Vb, const float* __restrict__ temp,
    float* __restrict__ Ctx)
{
    __shared__ unsigned pbufU[8 * 16 * 20];  // per-wave P tile, bf16 pairs

    const int tid  = threadIdx.x;
    const int wid  = tid >> 5;
    const int lane = tid & 31;
    const int half = lane >> 4;
    const int l16  = lane & 15;

    const int tileIdx = blockIdx.x * 8 + wid;        // 0..4095
    const int qt = tileIdx & 127;                    // L/16 = 128
    const int bh = tileIdx >> 7;
    const int b  = bh >> 3;
    const int h  = bh & 7;
    const int l0 = qt * 16;
    const float scale = 0.125f * temp[0];            // HD^-0.5 * temp folded into Q
    const size_t baseBL = (size_t)b * LL;
    unsigned* pw = &pbufU[wid * 16 * 20];
    unsigned short* pwh = (unsigned short*)pw;

    // per-(b,h) base pointers (hoisted)
    const float* Kbh = Kb + (baseBL * HH + h) * HD;  // + key*512 + feat
    const float* Vbh = Vb + (baseBL * HH + h) * HD;  // + key*512 + hd

    // Q fragments (2 K-chunks of 32 features), scale pre-folded
    v16bf qf[2];
    {
        const int row = l0 + l16;
        const float* qrow = Qb + ((baseBL + row) * HH + h) * HD;
#pragma unroll
        for (int ch = 0; ch < 2; ch++) {
            FragU f;
#pragma unroll
            for (int p = 0; p < 8; p++) {
                const int kk = ch * 32 + half * 16 + 2 * p;
                const float2 v = *(const float2*)(qrow + kk);
                f.u[p] = pack_bf16x2(v.x * scale, v.y * scale);
            }
            qf[ch] = f.v;
        }
    }

    v8f cacc[4];
#pragma unroll
    for (int j = 0; j < 4; j++) cacc[j] = zero_v8();
    float rs[8];
#pragma unroll
    for (int r = 0; r < 8; r++) rs[r] = 0.f;

    const int lmax = l0 + 15;
    for (int m0 = 0; m0 <= lmax; m0 += 32) {
        // ---- scores for two 16-key subtiles -> sigmoid -> pbuf (bf16) ----
#pragma unroll
        for (int sub = 0; sub < 2; sub++) {
            const int ms = m0 + sub * 16;
            const int key = ms + l16;
            const int keyc = key < LL ? key : LL - 1;   // clamp OOB (masked anyway)
            const float* krow = Kbh + (size_t)keyc * (HH * HD);
            v8f sacc = zero_v8();
#pragma unroll
            for (int ch = 0; ch < 2; ch++) {
                FragU kf;   // B 32x16: lane=key col, k=feature pairs
#pragma unroll
                for (int v = 0; v < 8; v++) {
                    const int kk = ch * 32 + half * 16 + 2 * v;
                    const float2 w = *(const float2*)(krow + kk);
                    kf.u[v] = pack_bf16x2(w.x, w.y);
                }
                sacc = __builtin_amdgcn_wmma_f32_16x16x32_bf16(
                    false, qf[ch], false, kf.v, (short)0, sacc, false, false);
            }
#pragma unroll
            for (int r = 0; r < 8; r++) {
                const int qrow = l0 + r + 8 * half;
                const float p = (key <= qrow) ? (1.f / (1.f + __expf(-sacc[r]))) : 0.f;
                rs[r] += p;
                pwh[(r + 8 * half) * 40 + sub * 16 + l16] = f2bf_bits(p);
            }
        }

        // ---- re-fragment P as A 16x32: 2x ds_load_b128 ----
        FragU pf;
        pf.q[0] = *(const uint4*)&pw[l16 * 20 + half * 8 + 0];
        pf.q[1] = *(const uint4*)&pw[l16 * 20 + half * 8 + 4];

        // ---- ctx += P(16x32) x V(32x64) : 4 HD-chunks ----
        const float* vbase = Vbh + (size_t)m0 * (HH * HD) + l16;
#pragma unroll
        for (int j = 0; j < 4; j++) {
            FragU vf;   // B 32x16: lane=hd col, k=key pairs (strided loads)
            const float* vb0 = vbase + j * 16;
#pragma unroll
            for (int v = 0; v < 8; v++) {
                const int kk = half * 16 + 2 * v;
                // keys beyond lmax are multiplied by P==0; clamp only the
                // final tile's overshoot to stay in-bounds
                const int koff = (m0 + kk + 1 < LL) ? kk : (LL - 2 - m0);
                const float x0 = vb0[(size_t)(koff)     * (HH * HD)];
                const float x1 = vb0[(size_t)(koff + 1) * (HH * HD)];
                vf.u[v] = pack_bf16x2(x0, x1);
            }
            cacc[j] = __builtin_amdgcn_wmma_f32_16x16x32_bf16(
                false, pf.v, false, vf.v, (short)0, cacc[j], false, false);
        }
    }

    // ---- rsum reduce (16-lane butterfly) and write ctx ----
#pragma unroll
    for (int r = 0; r < 8; r++) {
        float s = rs[r];
        s += __shfl_xor(s, 1, 16);
        s += __shfl_xor(s, 2, 16);
        s += __shfl_xor(s, 4, 16);
        s += __shfl_xor(s, 8, 16);
        rs[r] = 1.f / fmaxf(s, 1.f);
    }
#pragma unroll
    for (int j = 0; j < 4; j++) {
#pragma unroll
        for (int r = 0; r < 8; r++) {
            const int row = l0 + r + 8 * half;
            Ctx[((baseBL + row) * HH + h) * HD + j * 16 + l16] = cacc[j][r] * rs[r];
        }
    }
}

// =====================================================================
// Elementwise / norm helper kernels
// =====================================================================
__global__ __launch_bounds__(256) void encx_kernel(
    const float* __restrict__ vals, const float* __restrict__ Wi,
    const float* __restrict__ bi, float* __restrict__ X)
{
    const size_t i = (size_t)blockIdx.x * 256 + threadIdx.x;
    const int d = (int)(i & (DD - 1));
    const size_t m = i >> 9;
    X[i] = vals[m] * Wi[d] + bi[d];
}

// RMS norm: one wave per row of 512
__global__ __launch_bounds__(256) void rms_kernel(
    const float* __restrict__ X, const float* __restrict__ w, float* __restrict__ Y)
{
    const int row  = blockIdx.x * 8 + (threadIdx.x >> 5);
    const int lane = threadIdx.x & 31;
    const float* xr = X + (size_t)row * DD;
    float4 v[4];
    float ss = 0.f;
#pragma unroll
    for (int k = 0; k < 4; k++) {
        v[k] = *(const float4*)(xr + k * 128 + lane * 4);
        ss += v[k].x * v[k].x + v[k].y * v[k].y + v[k].z * v[k].z + v[k].w * v[k].w;
    }
    ss += __shfl_xor(ss, 1);  ss += __shfl_xor(ss, 2);
    ss += __shfl_xor(ss, 4);  ss += __shfl_xor(ss, 8);
    ss += __shfl_xor(ss, 16);
    const float sc = rsqrtf(ss * (1.f / DD) + 1e-6f);
    float* yr = Y + (size_t)row * DD;
#pragma unroll
    for (int k = 0; k < 4; k++) {
        const int c = k * 128 + lane * 4;
        yr[c + 0] = v[k].x * sc * w[c + 0];
        yr[c + 1] = v[k].y * sc * w[c + 1];
        yr[c + 2] = v[k].z * sc * w[c + 2];
        yr[c + 3] = v[k].w * sc * w[c + 3];
    }
}

// d2zn: second difference along L (per batch), L2-normalized
__global__ __launch_bounds__(256) void d2zn_kernel(
    const float* __restrict__ Z, float* __restrict__ Out)
{
    const int row  = blockIdx.x * 8 + (threadIdx.x >> 5);
    const int lane = threadIdx.x & 31;
    const int l = row & (LL - 1);
    const float* zl = Z + (size_t)row * DD;
    float d[16];
    float ss = 0.f;
#pragma unroll
    for (int k = 0; k < 4; k++) {
        const int c = k * 128 + lane * 4;
        const float4 a  = *(const float4*)(zl + c);
        const float4 b1 = (l >= 1) ? *(const float4*)(zl - DD + c) : a;
        const float4 b2 = (l >= 2) ? *(const float4*)(zl - 2 * DD + c) : b1;
        const float dz0x = (l >= 1) ? (a.x - b1.x) : 0.f;
        const float dz0y = (l >= 1) ? (a.y - b1.y) : 0.f;
        const float dz0z = (l >= 1) ? (a.z - b1.z) : 0.f;
        const float dz0w = (l >= 1) ? (a.w - b1.w) : 0.f;
        const float dppx = (l >= 2) ? (b1.x - b2.x) : 0.f;
        const float dppy = (l >= 2) ? (b1.y - b2.y) : 0.f;
        const float dppz = (l >= 2) ? (b1.z - b2.z) : 0.f;
        const float dppw = (l >= 2) ? (b1.w - b2.w) : 0.f;
        d[4 * k + 0] = dz0x - dppx;  d[4 * k + 1] = dz0y - dppy;
        d[4 * k + 2] = dz0z - dppz;  d[4 * k + 3] = dz0w - dppw;
        ss += d[4*k]*d[4*k] + d[4*k+1]*d[4*k+1] + d[4*k+2]*d[4*k+2] + d[4*k+3]*d[4*k+3];
    }
    ss += __shfl_xor(ss, 1);  ss += __shfl_xor(ss, 2);
    ss += __shfl_xor(ss, 4);  ss += __shfl_xor(ss, 8);
    ss += __shfl_xor(ss, 16);
    const float inv = 1.f / (sqrtf(ss) + 1e-6f);
    float* yr = Out + (size_t)row * DD;
#pragma unroll
    for (int k = 0; k < 4; k++) {
        const int c = k * 128 + lane * 4;
        yr[c + 0] = d[4*k + 0] * inv;  yr[c + 1] = d[4*k + 1] * inv;
        yr[c + 2] = d[4*k + 2] * inv;  yr[c + 3] = d[4*k + 3] * inv;
    }
}

__global__ __launch_bounds__(256) void dzmix_kernel(
    const float* __restrict__ dzl, const float* __restrict__ corr,
    const float* __restrict__ dt, float* __restrict__ dz)
{
    const size_t i = (size_t)blockIdx.x * 256 + threadIdx.x;
    dz[i] = dt[0] * (dzl[i] + corr[i]);
}

__global__ __launch_bounds__(256) void add3_kernel(
    float* __restrict__ z, const float* __restrict__ a, const float* __restrict__ bq)
{
    const size_t i = (size_t)blockIdx.x * 256 + threadIdx.x;
    z[i] = z[i] + a[i] + bq[i];
}

// prediction head on the last token of each batch (tiny, single WG)
__global__ __launch_bounds__(256) void pred_kernel(
    const float* __restrict__ zf, const float* __restrict__ vals,
    const float* __restrict__ W1, const float* __restrict__ b1,
    const float* __restrict__ W2, const float* __restrict__ b2,
    float* __restrict__ out)
{
    __shared__ float hbuf[DD];
    __shared__ float red[256];
    for (int b = 0; b < BB; b++) {
        const float* zr = zf + ((size_t)(b * LL + LL - 1)) * DD;
        for (int j = threadIdx.x; j < DD; j += 256) {
            float acc = b1[j];
            for (int k = 0; k < DD; k++) acc += zr[k] * W1[(size_t)k * DD + j];
            hbuf[j] = acc / (1.f + __expf(-acc));
        }
        __syncthreads();
        float p = 0.f;
        for (int j = threadIdx.x; j < DD; j += 256) p += hbuf[j] * W2[j];
        red[threadIdx.x] = p;
        __syncthreads();
        for (int s = 128; s > 0; s >>= 1) {
            if (threadIdx.x < s) red[threadIdx.x] += red[threadIdx.x + s];
            __syncthreads();
        }
        if (threadIdx.x == 0) {
            const float pd = red[0] + b2[0];
            const float xl = vals[b * LL + LL - 1];
            const float xp = vals[b * LL + LL - 2];
            out[b]     = xl + (xl - xp) + pd;   // x_new
            out[4 + b] = pd;                    // pred_d2x
        }
        __syncthreads();
    }
}

// =====================================================================
// Host orchestration
// =====================================================================
static inline void gemm(hipStream_t s,
                        const float* A0, const float* A1, const float* A2, const float* A3,
                        int lda, const float* W, int N, int K,
                        const float* bias, const float* res, float* C, int act)
{
    dim3 g(N / 128, MM / 128);
    gemm_wmma_kernel<<<g, 256, 0, s>>>(A0, A1, A2, A3, lda, W, N, K, bias, res, C, act);
}

extern "C" void kernel_launch(void* const* d_in, const int* in_sizes, int n_in,
                              void* d_out, int out_size, void* d_ws, size_t ws_size,
                              hipStream_t stream)
{
    (void)in_sizes; (void)n_in; (void)out_size; (void)ws_size;
    auto F = [&](int i) { return (const float*)d_in[i]; };

    // --- input layout: setup_inputs() insertion order ---
    const float* values = F(0);
    const float *Wi = F(1), *bi = F(2), *Wz = F(3), *bz = F(4);
    const float *Wc = F(5), *bc = F(6), *Wp = F(7), *bp = F(8);
    // blocks: base 9 + 25*blk, order:
    // wnz,wnc,wnm,Wf1,bf1,Wf2,bf2,Wg1,bg1,Wg2,bg2,dt,Wq,Wk,Wv,Wo,temp,
    // Wu1,bu1,Wu2,bu2,Wm1,bm1,Wm2,bm2
    const float* norm_w = F(9 + 2 * 25);
    const float *pW1 = F(60), *pb1 = F(61), *pW2 = F(62), *pb2 = F(63);

    float* out = (float*)d_out;
    float* zout = out + 8;

    const size_t SL = (size_t)MM * DD;  // one M x D slot
    float* ws = (float*)d_ws;
    float* S0 = ws + 0 * SL;   // z
    float* S1 = ws + 1 * SL;   // conn
    float* S2 = ws + 2 * SL;   // zn / znm
    float* S3 = ws + 3 * SL;   // cn / d2zn
    float* S4 = ws + 4 * SL;   // dzl / curv
    float* S5 = ws + 5 * SL;   // dz
    float* S6 = ws + 6 * SL;   // Q  | h1/u1 lo | m1 part
    float* S7 = ws + 7 * SL;   // K  | h1/u1 hi | m1 part
    float* S8 = ws + 8 * SL;   // V  | g        | m1 part
    float* S9 = ws + 9 * SL;   // ctx| corr     | m1 part
    float* T0 = ws + 10 * SL;  // ctxo

    const int EW = (int)(SL / 256);  // elementwise grid
    const int RW = MM / 8;           // row-wise grid (8 rows / WG)

    // ---------------- encoder ----------------
    encx_kernel<<<EW, 256, 0, stream>>>(values, Wi, bi, S2);              // x
    gemm(stream, S2, 0, 0, 0, DD, Wz, DD, DD, bz, 0, S0, 0);              // z
    d2zn_kernel<<<RW, 256, 0, stream>>>(S0, S3);                          // d2zn
    gemm(stream, S3, 0, 0, 0, DD, Wc, DD, DD, bc, 0, S4, 0);              // curv
    gemm(stream, S0, S4, 0, 0, DD, Wp, DD, 2 * DD, bp, 0, S1, 0);         // conn

    // ---------------- blocks ----------------
    for (int blk = 0; blk < 2; blk++) {
        const int o = 9 + blk * 25;
        const float *wnz = F(o+0), *wnc = F(o+1), *wnm = F(o+2);
        const float *Wf1 = F(o+3), *bf1 = F(o+4), *Wf2 = F(o+5), *bf2 = F(o+6);
        const float *Wg1 = F(o+7), *bg1 = F(o+8), *Wg2 = F(o+9), *bg2 = F(o+10);
        const float *dt  = F(o+11);
        const float *Wq  = F(o+12), *Wk = F(o+13), *Wv = F(o+14), *Wo = F(o+15);
        const float *temp = F(o+16);
        const float *Wu1 = F(o+17), *bu1 = F(o+18), *Wu2 = F(o+19), *bu2 = F(o+20);
        const float *Wm1 = F(o+21), *bm1 = F(o+22), *Wm2 = F(o+23), *bm2 = F(o+24);

        rms_kernel<<<RW, 256, 0, stream>>>(S0, wnz, S2);                  // zn
        rms_kernel<<<RW, 256, 0, stream>>>(S1, wnc, S3);                  // cn
        gemm(stream, S2, 0, 0, 0, DD, Wf1, 2*DD, DD, bf1, 0, S6, 1);      // h1=silu  (S6..S7)
        gemm(stream, S6, S6 + DD, 0, 0, 2*DD, Wf2, DD, 2*DD, bf2, 0, S4, 0); // dzl
        gemm(stream, S3, S4, 0, 0, DD, Wg1, DD, 2*DD, bg1, 0, S8, 2);     // g=tanh
        gemm(stream, S8, 0, 0, 0, DD, Wg2, DD, DD, bg2, 0, S9, 0);        // corr
        dzmix_kernel<<<EW, 256, 0, stream>>>(S4, S9, dt, S5);             // dz
        gemm(stream, S2, S3, 0, 0, DD, Wq, DD, 2*DD, 0, 0, S6, 0);        // Q
        gemm(stream, S2, S3, 0, 0, DD, Wk, DD, 2*DD, 0, 0, S7, 0);        // K
        gemm(stream, S2, 0, 0, 0, DD, Wv, DD, DD, 0, 0, S8, 0);           // V
        attn_kernel<<<(BB*HH*(LL/16))/8, 256, 0, stream>>>(S6, S7, S8, temp, S9); // ctx
        gemm(stream, S9, 0, 0, 0, DD, Wo, DD, DD, 0, 0, T0, 0);           // ctxo
        add3_kernel<<<EW, 256, 0, stream>>>(S0, S5, T0);                  // z += dz + ctxo
        gemm(stream, S1, S0, S5, 0, DD, Wu1, 2*DD, 3*DD, bu1, 0, S6, 1);  // u1=silu (S6..S7)
        gemm(stream, S6, S6 + DD, 0, 0, 2*DD, Wu2, DD, 2*DD, bu2, S1, S1, 0); // conn += upd
        rms_kernel<<<RW, 256, 0, stream>>>(S0, wnm, S2);                  // znm
        gemm(stream, S2, 0, 0, 0, DD, Wm1, 4*DD, DD, bm1, 0, S6, 1);      // m1=silu (S6..S9)
        gemm(stream, S6, S6 + DD, S6 + 2*DD, S6 + 3*DD, 4*DD,
             Wm2, DD, 4*DD, bm2, S0, S0, 0);                              // z += mlp
    }

    // ---------------- head ----------------
    rms_kernel<<<RW, 256, 0, stream>>>(S0, norm_w, zout);                 // z out (post-norm)
    pred_kernel<<<1, 256, 0, stream>>>(zout, values, pW1, pb1, pW2, pb2, out);
}
